// Next_Node_Probability_Calculator_for_group_21990232556010
// MI455X (gfx1250) — matrix operand order repaired and checked by reading, not compiled, not run on hardware
//
#include <hip/hip_runtime.h>
#include <hip/hip_bf16.h>

// ---------------------------------------------------------------------------
// MI455X / gfx1250 implementation.
//
// Precision strategy: all GEMM operands downconverted once to f16, f32
// accumulation via v_wmma_f32_16x16x32_f16 (codegen-confirmed builtin).
// Problem is HBM-bound (~50MB traffic vs ~15 GFLOP), so the two softmaxes are
// fused into the attention / pointer kernels through LDS so the [B,H,G,P] and
// [B,G,P] score tensors never touch HBM.
//
// Workspace usage: ~26 MB (f16 copies of enc/q/k/v, transposed f16 weights,
// f32 attention output).
// ---------------------------------------------------------------------------

#define BN 16    // batch
#define GN 512   // groups
#define PN 512   // pointers / nodes
#define EN 256   // embedding
#define HN 16    // heads
#define KN 16    // head dim

typedef __attribute__((ext_vector_type(16))) _Float16 v16h;
typedef __attribute__((ext_vector_type(8)))  float    v8f;

__device__ __forceinline__ v8f vzero8() {
  v8f c;
#pragma unroll
  for (int i = 0; i < 8; ++i) c[i] = 0.0f;
  return c;
}

__device__ __forceinline__ v8f wmma_f16(v16h a, v16h b, v8f c) {
  // D = A(16x32 f16) * B(32x16 f16) + C(16x16 f32)
  return __builtin_amdgcn_wmma_f32_16x16x32_f16(
      /*neg_a=*/false, a, /*neg_b=*/false, b,
      /*c_mod=*/(short)0, c, /*reuse_a=*/false, /*reuse_b=*/false);
}

// A-tile load from row-major f32 source. `p` points at (row l16, col k0).
// A layout (contiguous, matching B): lane half h: elements 0..15 hold
// K = 16h+0..15.
__device__ __forceinline__ v16h a_from_f32(const float* p, int half) {
  v16h a;
#pragma unroll
  for (int i = 0; i < 8; ++i) {
    a[i]     = (_Float16)p[16 * half + i];
    a[i + 8] = (_Float16)p[16 * half + 8 + i];
  }
  return a;
}

// ---------------------------------------------------------------------------
// Prep kernels
// ---------------------------------------------------------------------------

// Transpose the four 256x256 weight blocks to [out][in] f16 so WMMA B-tiles
// (lane = out column, contiguous K run per lane-half) are vector loads.
__global__ void prep_weights_kernel(const float* __restrict__ Wq,
                                    const float* __restrict__ Wk,
                                    const float* __restrict__ Wv,
                                    const float* __restrict__ Wc,
                                    _Float16* __restrict__ wq2t,
                                    _Float16* __restrict__ wkt,
                                    _Float16* __restrict__ wvt,
                                    _Float16* __restrict__ wct) {
  int tid   = blockIdx.x * blockDim.x + threadIdx.x;  // 4*256*256 threads
  int which = tid >> 16;
  int rem   = tid & 65535;          // = n*256 + k
  int n = rem >> 8, k = rem & 255;
  float v;
  _Float16* dst;
  switch (which) {
    case 0:  v = Wk[k * 256 + n];           dst = wkt;  break;
    case 1:  v = Wv[k * 256 + n];           dst = wvt;  break;
    case 2:  v = Wq[(EN + k) * 256 + n];    dst = wq2t; break;  // rows E..2E-1
    default: v = Wc[k * 256 + n];           dst = wct;  break;
  }
  dst[rem] = (_Float16)v;
}

__global__ void f32_to_f16_kernel(const float* __restrict__ src,
                                  _Float16* __restrict__ dst, int n) {
  int i = blockIdx.x * blockDim.x + threadIdx.x;
  if (i < n) dst[i] = (_Float16)src[i];
}

// q1[b,n] = input1[b,:] . Wq[0:E, n]   (g-invariant part of the concat GEMM)
__global__ void q1_kernel(const float* __restrict__ in1,
                          const float* __restrict__ Wq,
                          float* __restrict__ q1) {
  int tid = blockIdx.x * blockDim.x + threadIdx.x;  // B*256
  int b = tid >> 8, n = tid & 255;
  float s = 0.0f;
  for (int e = 0; e < EN; ++e) s += in1[b * EN + e] * Wq[e * 256 + n];
  q1[tid] = s;
}

// ---------------------------------------------------------------------------
// K / V projection: one wave computes one (b, p-tile, head) 16x16 output for
// both K and V (shared A tile).  K stored [B,H,P,16], V stored [B,H,16,P].
// ---------------------------------------------------------------------------
__global__ void __launch_bounds__(128) proj_kv_kernel(
    const float* __restrict__ enc, const _Float16* __restrict__ wkt,
    const _Float16* __restrict__ wvt, _Float16* __restrict__ kh,
    _Float16* __restrict__ vT) {
  int wave = (blockIdx.x * blockDim.x + threadIdx.x) >> 5;  // 0..8191
  int lane = threadIdx.x & 31;
  int l16 = lane & 15, half = lane >> 4;
  int h  = wave & 15;
  int pt = (wave >> 4) & 31;
  int b  = wave >> 9;

  v8f ck = vzero8(), cv = vzero8();
  for (int k0 = 0; k0 < EN; k0 += 32) {
    const float* ap = enc + ((size_t)b * PN + pt * 16 + l16) * EN + k0;
    v16h a  = a_from_f32(ap, half);
    v16h bk = *(const v16h*)(wkt + (h * 16 + l16) * 256 + k0 + 16 * half);
    v16h bv = *(const v16h*)(wvt + (h * 16 + l16) * 256 + k0 + 16 * half);
    ck = wmma_f16(a, bk, ck);
    cv = wmma_f16(a, bv, cv);
  }
  int bh = b * HN + h;
#pragma unroll
  for (int r = 0; r < 8; ++r) {
    int p = pt * 16 + r + 8 * half;
    kh[((size_t)bh * PN + p) * 16 + l16] = (_Float16)ck[r];
    vT[((size_t)bh * 16 + l16) * PN + p] = (_Float16)cv[r];
  }
}

// ---------------------------------------------------------------------------
// Q projection: input2 @ Wq[E:2E] + q1[b] + t[b,g]*Wq[2E].  Stored [B,H,G,16].
// ---------------------------------------------------------------------------
__global__ void __launch_bounds__(128) qproj_kernel(
    const float* __restrict__ in2, const float* __restrict__ Wq,
    const float* __restrict__ q1, const float* __restrict__ t,
    const _Float16* __restrict__ wq2t, _Float16* __restrict__ qh) {
  int wave = (blockIdx.x * blockDim.x + threadIdx.x) >> 5;
  int lane = threadIdx.x & 31;
  int l16 = lane & 15, half = lane >> 4;
  int h  = wave & 15;
  int gt = (wave >> 4) & 31;
  int b  = wave >> 9;

  v8f c = vzero8();
  for (int k0 = 0; k0 < EN; k0 += 32) {
    const float* ap = in2 + ((size_t)b * GN + gt * 16 + l16) * EN + k0;
    v16h a  = a_from_f32(ap, half);
    v16h bt = *(const v16h*)(wq2t + (h * 16 + l16) * 256 + k0 + 16 * half);
    c = wmma_f16(a, bt, c);
  }
  const float* wlast = Wq + (size_t)(2 * EN) * 256;  // row 2E
#pragma unroll
  for (int r = 0; r < 8; ++r) {
    int g = gt * 16 + r + 8 * half;
    int n = h * 16 + l16;
    float val = c[r] + q1[b * 256 + n] + t[(size_t)b * GN + g] * wlast[n];
    qh[((size_t)(b * HN + h) * GN + g) * 16 + l16] = (_Float16)val;
  }
}

// ---------------------------------------------------------------------------
// Fused attention: one wave per (b, h, g-tile).  Head dim (16) is zero-padded
// to WMMA K=32 on both operands.  Scores [16x512] live in LDS; two-pass
// softmax; second WMMA pass contracts exp-weights against V over P.
// ---------------------------------------------------------------------------
__global__ void __launch_bounds__(32) attn_kernel(
    const _Float16* __restrict__ qh, const _Float16* __restrict__ kh,
    const _Float16* __restrict__ vT, const float* __restrict__ mask,
    float* __restrict__ outc) {
  __shared__ float s_s[16 * PN];  // 32 KB score tile
  __shared__ float s_m[16];
  __shared__ float s_l[16];

  int lane = threadIdx.x & 31;
  int l16 = lane & 15, half = lane >> 4;
  int gt = blockIdx.x & 31;
  int h  = (blockIdx.x >> 5) & 15;
  int b  = blockIdx.x >> 9;
  int bh = b * HN + h;

  // A(q) tile loaded once: K>=16 padded with zero.
  v16h aq;
  {
    const _Float16* qrow = qh + ((size_t)bh * GN + gt * 16 + l16) * 16;
#pragma unroll
    for (int i = 0; i < 16; ++i) aq[i] = half ? (_Float16)0.0f : qrow[i];
  }

  // Pass 1: scores = q.k / sqrt(16) + mask
  for (int p0 = 0; p0 < PN; p0 += 16) {
    v16h bk;
#pragma unroll
    for (int i = 0; i < 16; ++i) bk[i] = (_Float16)0.0f;   // K=16..31 pad
    if (half == 0)
      bk = *(const v16h*)(kh + ((size_t)bh * PN + p0 + l16) * 16);
    v8f c = wmma_f16(aq, bk, vzero8());
#pragma unroll
    for (int r = 0; r < 8; ++r) {
      int M = r + 8 * half;
      float s = c[r] * 0.25f +
                mask[((size_t)b * GN + gt * 16 + M) * PN + p0 + l16];
      s_s[M * PN + p0 + l16] = s;
    }
  }
  __syncthreads();

  // Row max / sum, two lanes per row.
  {
    int row = lane >> 1, par = lane & 1;
    float m = -1e30f;
    for (int j = par; j < PN; j += 2) m = fmaxf(m, s_s[row * PN + j]);
    m = fmaxf(m, __shfl_xor(m, 1, 32));
    float l = 0.0f;
    for (int j = par; j < PN; j += 2) l += __expf(s_s[row * PN + j] - m);
    l += __shfl_xor(l, 1, 32);
    if (par == 0) { s_m[row] = m; s_l[row] = l; }
  }
  __syncthreads();

  // Pass 2: O = exp(s-m) @ V, contraction over P in chunks of 32.
  v8f co = vzero8();
  for (int p0 = 0; p0 < PN; p0 += 32) {
    float m = s_m[l16];
    v16h aw;
#pragma unroll
    for (int i = 0; i < 8; ++i) {
      aw[i]     = (_Float16)__expf(s_s[l16 * PN + p0 + 16 * half + i] - m);
      aw[i + 8] = (_Float16)__expf(s_s[l16 * PN + p0 + 16 * half + 8 + i] - m);
    }
    v16h bv = *(const v16h*)(vT + ((size_t)bh * 16 + l16) * PN + p0 + 16 * half);
    co = wmma_f16(aw, bv, co);
  }
#pragma unroll
  for (int r = 0; r < 8; ++r) {
    int M = r + 8 * half;
    int g = gt * 16 + M;
    outc[((size_t)b * GN + g) * 256 + h * 16 + l16] = co[r] / s_l[M];
  }
}

// ---------------------------------------------------------------------------
// Fused output projection + pointer softmax: one wave per (b, g-tile).
// mh tile (16x256, f16) built in LDS; then scores2 = mh @ enc^T / 16,
// 10*tanh + mask, softmax over P, stream probs to HBM.
// ---------------------------------------------------------------------------
__global__ void __launch_bounds__(32) final_kernel(
    const float* __restrict__ outc, const _Float16* __restrict__ wct,
    const float* __restrict__ Wc_b, const _Float16* __restrict__ ehalf,
    const float* __restrict__ mask, float* __restrict__ probs) {
  __shared__ _Float16 s_mh[16 * 256];  // 8 KB
  __shared__ float    s_s[16 * PN];    // 32 KB
  __shared__ float    s_m[16], s_l[16];

  int lane = threadIdx.x & 31;
  int l16 = lane & 15, half = lane >> 4;
  int gt = blockIdx.x & 31;
  int b  = blockIdx.x >> 5;

  // Step 1: mh = outc_tile @ Wc_w + bias  (into LDS, f16)
  for (int n0 = 0; n0 < 256; n0 += 16) {
    v8f c = vzero8();
    for (int k0 = 0; k0 < 256; k0 += 32) {
      const float* ap = outc + ((size_t)b * GN + gt * 16 + l16) * 256 + k0;
      v16h a  = a_from_f32(ap, half);
      v16h bt = *(const v16h*)(wct + (n0 + l16) * 256 + k0 + 16 * half);
      c = wmma_f16(a, bt, c);
    }
#pragma unroll
    for (int r = 0; r < 8; ++r) {
      int M = r + 8 * half;
      int n = n0 + l16;
      s_mh[M * 256 + n] = (_Float16)(c[r] + Wc_b[n]);
    }
  }
  __syncthreads();

  // Step 2: scores2 = 10*tanh(mh @ enc^T / sqrt(E)) + mask
  for (int p0 = 0; p0 < PN; p0 += 16) {
    v8f c = vzero8();
    for (int e0 = 0; e0 < EN; e0 += 32) {
      v16h a;
#pragma unroll
      for (int i = 0; i < 8; ++i) {
        a[i]     = s_mh[l16 * 256 + e0 + 16 * half + i];
        a[i + 8] = s_mh[l16 * 256 + e0 + 16 * half + 8 + i];
      }
      v16h bt = *(const v16h*)(ehalf + ((size_t)b * PN + p0 + l16) * EN +
                               e0 + 16 * half);
      c = wmma_f16(a, bt, c);
    }
#pragma unroll
    for (int r = 0; r < 8; ++r) {
      int M = r + 8 * half;
      float s2 = 10.0f * tanhf(c[r] * 0.0625f) +
                 mask[((size_t)b * GN + gt * 16 + M) * PN + p0 + l16];
      s_s[M * PN + p0 + l16] = s2;
    }
  }
  __syncthreads();

  {
    int row = lane >> 1, par = lane & 1;
    float m = -1e30f;
    for (int j = par; j < PN; j += 2) m = fmaxf(m, s_s[row * PN + j]);
    m = fmaxf(m, __shfl_xor(m, 1, 32));
    float l = 0.0f;
    for (int j = par; j < PN; j += 2) l += __expf(s_s[row * PN + j] - m);
    l += __shfl_xor(l, 1, 32);
    if (par == 0) { s_m[row] = m; s_l[row] = l; }
  }
  __syncthreads();

  for (int j = lane; j < 16 * PN; j += 32) {
    int row = j >> 9, p = j & (PN - 1);
    probs[((size_t)b * GN + gt * 16 + row) * PN + p] =
        __expf(s_s[j] - s_m[row]) / s_l[row];
  }
}

// ---------------------------------------------------------------------------
// Launch
// ---------------------------------------------------------------------------
extern "C" void kernel_launch(void* const* d_in, const int* in_sizes, int n_in,
                              void* d_out, int out_size, void* d_ws,
                              size_t ws_size, hipStream_t stream) {
  (void)in_sizes; (void)n_in; (void)out_size; (void)ws_size;

  const float* in1  = (const float*)d_in[0];  // [B,1,E]
  const float* in2  = (const float*)d_in[1];  // [B,G,E]
  const float* t    = (const float*)d_in[2];  // [B,G,1]
  const float* mask = (const float*)d_in[3];  // [B,G,P]
  const float* enc  = (const float*)d_in[4];  // [B,P,E]
  const float* Wq   = (const float*)d_in[5];  // [2E+1, 256]
  const float* Wk   = (const float*)d_in[6];  // [E, 256]
  const float* Wv   = (const float*)d_in[7];  // [E, 256]
  const float* Wc   = (const float*)d_in[8];  // [256, E]
  const float* Wcb  = (const float*)d_in[9];  // [E]
  float* probs = (float*)d_out;               // [B,G,P]

  // Workspace carve-out (~26 MB).
  char* ws = (char*)d_ws;
  size_t off = 0;
  auto take = [&](size_t bytes) {
    size_t r = off;
    off += (bytes + 255) & ~(size_t)255;
    return r;
  };
  _Float16* wkt   = (_Float16*)(ws + take((size_t)256 * 256 * 2));
  _Float16* wvt   = (_Float16*)(ws + take((size_t)256 * 256 * 2));
  _Float16* wq2t  = (_Float16*)(ws + take((size_t)256 * 256 * 2));
  _Float16* wct   = (_Float16*)(ws + take((size_t)256 * 256 * 2));
  float*    q1    = (float*)   (ws + take((size_t)BN * 256 * 4));
  _Float16* ehalf = (_Float16*)(ws + take((size_t)BN * PN * EN * 2));
  _Float16* qh    = (_Float16*)(ws + take((size_t)BN * HN * GN * 16 * 2));
  _Float16* kh    = (_Float16*)(ws + take((size_t)BN * HN * PN * 16 * 2));
  _Float16* vT    = (_Float16*)(ws + take((size_t)BN * HN * 16 * PN * 2));
  float*    outc  = (float*)   (ws + take((size_t)BN * GN * 256 * 4));

  // 1) weight transpose/convert + enc f16 + q1
  prep_weights_kernel<<<(4 * 256 * 256) / 256, 256, 0, stream>>>(
      Wq, Wk, Wv, Wc, wq2t, wkt, wvt, wct);
  f32_to_f16_kernel<<<(BN * PN * EN) / 256, 256, 0, stream>>>(
      enc, ehalf, BN * PN * EN);
  q1_kernel<<<(BN * 256) / 256, 256, 0, stream>>>(in1, Wq, q1);

  // 2) K/V projections  (B * P/16 * H = 8192 waves, 4 waves/block)
  proj_kv_kernel<<<(BN * (PN / 16) * HN) / 4, 128, 0, stream>>>(
      enc, wkt, wvt, kh, vT);

  // 3) Q projection
  qproj_kernel<<<(BN * (GN / 16) * HN) / 4, 128, 0, stream>>>(
      in2, Wq, q1, t, wq2t, qh);

  // 4) fused attention (one wave per (b,h,g-tile))
  attn_kernel<<<BN * HN * (GN / 16), 32, 0, stream>>>(qh, kh, vT, mask, outc);

  // 5) fused output projection + pointer softmax
  final_kernel<<<BN * (GN / 16), 32, 0, stream>>>(outc, wct, Wcb, ehalf, mask,
                                                  probs);
}
